// TFXLNetMainLayer_16106127360560
// MI455X (gfx1250) — matrix operand order repaired and checked by reading, not compiled, not run on hardware
//
#include <hip/hip_runtime.h>
#include <cstdint>

typedef __bf16 bf16;
typedef __attribute__((ext_vector_type(16))) __bf16 v16bf;
typedef __attribute__((ext_vector_type(8)))  __bf16 v8bf;
typedef __attribute__((ext_vector_type(8)))  float  v8f;

#define LDSW 40  // LDS row stride in bf16 elems (32 data + 8 pad) => 80B rows, 16B-aligned

__device__ __forceinline__ v16bf cat16(v8bf lo, v8bf hi) {
  return __builtin_shufflevector(lo, hi, 0, 1, 2, 3, 4, 5, 6, 7, 8, 9, 10, 11, 12, 13, 14, 15);
}

// Async DMA: global -> LDS, 16 bytes per lane, tracked by ASYNCcnt.
__device__ __forceinline__ void async_copy_b128(uint32_t lds_byte_addr, const void* gaddr) {
  asm volatile("global_load_async_to_lds_b128 %0, %1, off"
               :: "v"(lds_byte_addr), "v"(gaddr)
               : "memory");
}
__device__ __forceinline__ void wait_async0() {
  asm volatile("s_wait_asynccnt 0x0" ::: "memory");
}

// ---------------------------------------------------------------------------
// NT GEMM: C = A[M,K] * B^T, B stored row-major [N,K]. bf16 in, f32 accumulate,
// V_WMMA_F32_16X16X32_BF16. Block 128x128, BK=32, 256 threads (8 waves),
// wave tile 32x64 (2x4 WMMA). Double-buffered LDS fed by async global->LDS
// copies; one barrier per K-step.
//   SHIFTED == 1: rel-shift store  C[p, q-(N-Nout)+p] += val   (bd term)
// Requires: M % 128 == 0, K % 32 == 0, K >= 64; OOB N rows may be read (must be
// mapped memory) but never stored (col < Nout guard).
// ---------------------------------------------------------------------------
template <int SHIFTED>
__global__ __launch_bounds__(256) void gemm_nt_wmma(
    const bf16* __restrict__ A, const bf16* __restrict__ B, float* __restrict__ C,
    int M, int N, int K, int Nout, int lda, int ldb, int ldc,
    long long sA, long long sB, long long sC) {
  const int m0 = blockIdx.y * 128;
  const int n0 = blockIdx.x * 128;
  if (SHIFTED) {
    const int shift = N - Nout;                 // cs = col - shift + row
    if (n0 + m0 + 254 < shift) return;          // tile fully left of band
    if (n0 - shift + m0 >= Nout) return;        // tile fully right of band
  }
  A += (long long)blockIdx.z * sA;
  B += (long long)blockIdx.z * sB;
  C += (long long)blockIdx.z * sC;

  __shared__ __align__(16) bf16 As[2][128 * LDSW];
  __shared__ __align__(16) bf16 Bs[2][128 * LDSW];

  const int tid  = threadIdx.x;
  const int lane = tid & 31;
  const int wid  = tid >> 5;
  const int wm   = wid >> 1;   // 0..3 : 32-row wave block
  const int wn   = wid & 1;    // 0..1 : 64-col wave block
  const int half = lane >> 4;
  const int l16  = lane & 15;

  // staging geometry: 512 16B-chunks per tile, 2 per thread per matrix
  const int r_st = tid >> 2;          // 0..63 (+64 for second chunk)
  const int k_st = (tid & 3) * 8;     // 0,8,16,24 (bf16 elems)

  auto stage = [&](int buf, int kk) {
#pragma unroll
    for (int c = 0; c < 2; ++c) {
      int ar = r_st + c * 64;
      uint32_t lds = (uint32_t)(uintptr_t)(&As[buf][ar * LDSW + k_st]);
      async_copy_b128(lds, A + (long long)(m0 + ar) * lda + (kk + k_st));
    }
#pragma unroll
    for (int c = 0; c < 2; ++c) {
      int br = r_st + c * 64;
      uint32_t lds = (uint32_t)(uintptr_t)(&Bs[buf][br * LDSW + k_st]);
      async_copy_b128(lds, B + (long long)(n0 + br) * ldb + (kk + k_st));
    }
  };

  v8f acc[2][4];
#pragma unroll
  for (int mi = 0; mi < 2; ++mi)
#pragma unroll
    for (int ni = 0; ni < 4; ++ni)
#pragma unroll
      for (int e = 0; e < 8; ++e) acc[mi][ni][e] = 0.0f;

  stage(0, 0);
  wait_async0();
  __syncthreads();

  const int nk = K >> 5;
  for (int t = 0; t < nk; ++t) {
    const int buf = t & 1;
    if (t + 1 < nk) stage(buf ^ 1, (t + 1) << 5);  // async prefetch next tile

    // fragments per ISA 7.12.2 (A: lane row=l16, lo=K[8h..], hi=K[16+8h..];
    // B from [n][k]-major LDS: lane col=l16, lo=K[8h..], hi=K[16+8h..])
    v16bf af[2], bfg[4];
#pragma unroll
    for (int mi = 0; mi < 2; ++mi) {
      const bf16* p = &As[buf][(wm * 32 + mi * 16 + l16) * LDSW];
      v8bf lo = *(const v8bf*)(p + half * 8);
      v8bf hi = *(const v8bf*)(p + 16 + half * 8);
      af[mi] = cat16(lo, hi);
    }
#pragma unroll
    for (int ni = 0; ni < 4; ++ni) {
      const bf16* p = &Bs[buf][(wn * 64 + ni * 16 + l16) * LDSW];
      v8bf lo = *(const v8bf*)(p + half * 8);
      v8bf hi = *(const v8bf*)(p + 16 + half * 8);
      bfg[ni] = cat16(lo, hi);
    }
#pragma unroll
    for (int mi = 0; mi < 2; ++mi)
#pragma unroll
      for (int ni = 0; ni < 4; ++ni)
        acc[mi][ni] = __builtin_amdgcn_wmma_f32_16x16x32_bf16(
            false, af[mi], false, bfg[ni], (short)0, acc[mi][ni], false, false);

    wait_async0();     // own async copies for next tile complete
    __syncthreads();   // everyone's copies visible; prev buffer free
  }

  const int shift = SHIFTED ? (N - Nout) : 0;
#pragma unroll
  for (int mi = 0; mi < 2; ++mi) {
    int r0 = m0 + wm * 32 + mi * 16;
#pragma unroll
    for (int ni = 0; ni < 4; ++ni) {
      int c0 = n0 + wn * 64 + ni * 16;
#pragma unroll
      for (int gi = 0; gi < 8; ++gi) {
        int row = r0 + gi + 8 * half;
        int col = c0 + l16;
        float v = acc[mi][ni][gi];
        if (SHIFTED) {
          int cs = col - shift + row;  // fused rel_shift: bd[i,c] = raw[i, c+shift-i]
          if (cs >= 0 && cs < Nout) C[(long long)row * ldc + cs] += v;
        } else {
          if (col < Nout) C[(long long)row * ldc + col] = v;
        }
      }
    }
  }
}

// ---------------------------------------------------------------------------
// Elementwise / transpose / reduction kernels
// ---------------------------------------------------------------------------
__global__ void k_cast_bf16(const float* __restrict__ in, bf16* __restrict__ out, long long n) {
  long long i = (long long)blockIdx.x * blockDim.x + threadIdx.x;
  if (i < n) out[i] = (bf16)in[i];
}

// out[c][r] = (bf16)in[r][c]; grid (cols/32, rows/32), block (32,8)
__global__ void k_transpose_cast(const float* __restrict__ in, bf16* __restrict__ out,
                                 int rows, int cols) {
  __shared__ bf16 t[32][33];
  int bx = blockIdx.x * 32, by = blockIdx.y * 32;
#pragma unroll
  for (int i = 0; i < 32; i += 8) {
    int r = by + threadIdx.y + i, c = bx + threadIdx.x;
    if (r < rows && c < cols) t[threadIdx.y + i][threadIdx.x] = (bf16)in[(long long)r * cols + c];
  }
  __syncthreads();
#pragma unroll
  for (int i = 0; i < 32; i += 8) {
    int orow = bx + threadIdx.y + i, ocol = by + threadIdx.x;
    if (orow < cols && ocol < rows)
      out[(long long)orow * rows + ocol] = t[threadIdx.x][threadIdx.y + i];
  }
}

// vt[z][d][j] = v[j][z*64+d]  (v row stride 2048); grid (32, 2, 32), block (32,8)
__global__ void k_transpose_v(const bf16* __restrict__ v, bf16* __restrict__ vt) {
  __shared__ bf16 t[32][33];
  const int z = blockIdx.z;
  const int j0 = blockIdx.x * 32, d0 = blockIdx.y * 32;
#pragma unroll
  for (int i = 0; i < 32; i += 8)
    t[threadIdx.y + i][threadIdx.x] =
        v[(long long)(j0 + threadIdx.y + i) * 2048 + z * 64 + d0 + threadIdx.x];
  __syncthreads();
#pragma unroll
  for (int i = 0; i < 32; i += 8)
    vt[(long long)z * 65536 + (long long)(d0 + threadIdx.y + i) * 1024 + j0 + threadIdx.x] =
        t[threadIdx.x][threadIdx.y + i];
}

__global__ void k_addbias_cast(const float* __restrict__ in, const float* __restrict__ bias,
                               int period, bf16* __restrict__ out, long long n) {
  long long i = (long long)blockIdx.x * blockDim.x + threadIdx.x;
  if (i < n) out[i] = (bf16)(in[i] + bias[(int)(i % period)]);
}

__global__ void k_bias_gelu_cast(const float* __restrict__ in, const float* __restrict__ bias,
                                 int period, bf16* __restrict__ out, long long n) {
  long long i = (long long)blockIdx.x * blockDim.x + threadIdx.x;
  if (i < n) {
    float x = in[i] + bias[(int)(i % period)];
    float y = 0.5f * x * (1.0f + erff(x * 0.70710678118654752f));  // exact gelu
    out[i] = (bf16)y;
  }
}

// ef1[i,b,n,s] = sum_d (q[i,b,n,d] + r_s_bias[n,d]) * seg_embed[s,n,d]
__global__ void k_ef1(const float* __restrict__ q, const float* __restrict__ r_s_bias,
                      const float* __restrict__ seg_embed, float* __restrict__ ef1, int total) {
  int idx = blockIdx.x * blockDim.x + threadIdx.x;  // over S*B*H
  if (idx >= total) return;
  int n = idx & 15;
  int ib = idx >> 4;
  const float* qp = q + (long long)ib * 1024 + n * 64;
  const float* bp = r_s_bias + n * 64;
  const float* s0p = seg_embed + n * 64;
  const float* s1p = seg_embed + 1024 + n * 64;
  float s0 = 0.f, s1 = 0.f;
#pragma unroll 8
  for (int d = 0; d < 64; ++d) {
    float qv = qp[d] + bp[d];
    s0 += qv * s0p[d];
    s1 += qv * s1p[d];
  }
  ef1[(long long)idx * 2 + 0] = s0;
  ef1[(long long)idx * 2 + 1] = s1;
}

// scores[z,i,:] (z = b*16+n) holds ac+bd. Add segment term + mask, scale,
// softmax over keys, emit bf16 probabilities.
__global__ __launch_bounds__(256) void k_softmax(
    const float* __restrict__ scores, const float* __restrict__ ef1,
    const float* __restrict__ seg_mat, const float* __restrict__ mask,
    bf16* __restrict__ prob) {
  const int i = blockIdx.x;
  const int z = blockIdx.y;
  const int b = z >> 4;
  const int n = z & 15;
  const float* srow = scores + ((long long)z * 1024 + i) * 1024;
  bf16* prow = prob + ((long long)z * 1024 + i) * 1024;
  const float e0 = ef1[(((long long)i * 2 + b) * 16 + n) * 2 + 0];
  const float e1 = ef1[(((long long)i * 2 + b) * 16 + n) * 2 + 1];
  __shared__ float red[256];
  float vals[4];
  float mx = -3.0e38f;
#pragma unroll
  for (int t = 0; t < 4; ++t) {
    int c = threadIdx.x + t * 256;
    long long qc = (long long)i * 1024 + c;
    float m  = mask[qc * 2 + b];                  // [q,k,B,1]
    float s0 = seg_mat[(qc * 2 + b) * 2 + 0];     // [q,k,B,2]
    float s1 = seg_mat[(qc * 2 + b) * 2 + 1];
    float v = (srow[c] + s0 * e0 + s1 * e1) * 0.125f - 1.0e30f * m;
    vals[t] = v;
    mx = fmaxf(mx, v);
  }
  red[threadIdx.x] = mx;
  __syncthreads();
  for (int s = 128; s > 0; s >>= 1) {
    if (threadIdx.x < s) red[threadIdx.x] = fmaxf(red[threadIdx.x], red[threadIdx.x + s]);
    __syncthreads();
  }
  mx = red[0];
  __syncthreads();
  float sum = 0.f;
#pragma unroll
  for (int t = 0; t < 4; ++t) {
    vals[t] = __expf(vals[t] - mx);
    sum += vals[t];
  }
  red[threadIdx.x] = sum;
  __syncthreads();
  for (int s = 128; s > 0; s >>= 1) {
    if (threadIdx.x < s) red[threadIdx.x] += red[threadIdx.x + s];
    __syncthreads();
  }
  float inv = 1.0f / red[0];
#pragma unroll
  for (int t = 0; t < 4; ++t) prow[threadIdx.x + t * 256] = (bf16)(vals[t] * inv);
}

// out = LayerNorm(x + resid + optional bias) * gamma + beta  (cols == 1024)
__global__ __launch_bounds__(256) void k_resid_ln(
    const float* __restrict__ x, const float* __restrict__ resid,
    const float* __restrict__ bias, const float* __restrict__ gamma,
    const float* __restrict__ beta, float* __restrict__ out_f,
    bf16* __restrict__ out_bf, int cols) {
  const long long row = blockIdx.x;
  const float* xr = x + row * cols;
  const float* rr = resid + row * cols;
  __shared__ float red[256];
  float loc[4];
  const int nper = cols >> 8;  // 4
  float s = 0.f;
  for (int t = 0; t < nper; ++t) {
    int c = threadIdx.x + (t << 8);
    float v = xr[c] + rr[c] + (bias ? bias[c] : 0.0f);
    loc[t] = v;
    s += v;
  }
  red[threadIdx.x] = s;
  __syncthreads();
  for (int st = 128; st > 0; st >>= 1) {
    if (threadIdx.x < st) red[threadIdx.x] += red[threadIdx.x + st];
    __syncthreads();
  }
  float mean = red[0] / (float)cols;
  __syncthreads();
  float vs = 0.f;
  for (int t = 0; t < nper; ++t) {
    float d = loc[t] - mean;
    vs += d * d;
  }
  red[threadIdx.x] = vs;
  __syncthreads();
  for (int st = 128; st > 0; st >>= 1) {
    if (threadIdx.x < st) red[threadIdx.x] += red[threadIdx.x + st];
    __syncthreads();
  }
  float rstd = rsqrtf(red[0] / (float)cols + 1e-12f);
  for (int t = 0; t < nper; ++t) {
    int c = threadIdx.x + (t << 8);
    float y = (loc[t] - mean) * rstd * gamma[c] + beta[c];
    out_f[row * cols + c] = y;
    if (out_bf) out_bf[row * cols + c] = (bf16)y;
  }
}

// ---------------------------------------------------------------------------
// Orchestration
// ---------------------------------------------------------------------------
extern "C" void kernel_launch(void* const* d_in, const int* in_sizes, int n_in,
                              void* d_out, int out_size, void* d_ws, size_t ws_size,
                              hipStream_t stream) {
  (void)in_sizes; (void)n_in; (void)out_size; (void)ws_size;
  const int S = 1024, H = 16, D = 64, DM = 1024, DI = 4096, R = 2048;
  const int SB = 2048;          // S*B rows for [i,b] flattening
  const int RB = 4096;          // R*B
  const long long SS = (long long)S * S;

  const float* h        = (const float*)d_in[0];
  const float* g        = (const float*)d_in[1];
  const float* r        = (const float*)d_in[2];
  const float* mask_h   = (const float*)d_in[3];
  const float* mask_g   = (const float*)d_in[4];
  const float* seg_mat  = (const float*)d_in[5];
  const float* wq       = (const float*)d_in[6];
  const float* wk       = (const float*)d_in[7];
  const float* wv       = (const float*)d_in[8];
  const float* wo       = (const float*)d_in[9];
  const float* wr       = (const float*)d_in[10];
  const float* r_w_bias = (const float*)d_in[11];
  const float* r_r_bias = (const float*)d_in[12];
  const float* r_s_bias = (const float*)d_in[13];
  const float* seg_emb  = (const float*)d_in[14];
  const float* ln_g_at  = (const float*)d_in[15];
  const float* ln_b_at  = (const float*)d_in[16];
  const float* ff_w1    = (const float*)d_in[17];
  const float* ff_b1    = (const float*)d_in[18];
  const float* ff_w2    = (const float*)d_in[19];
  const float* ff_b2    = (const float*)d_in[20];
  const float* ln_g_ff  = (const float*)d_in[21];
  const float* ln_b_ff  = (const float*)d_in[22];
  float* out = (float*)d_out;

  // ---- bump allocator over workspace ----
  char* wsp = (char*)d_ws;
  auto alloc = [&](size_t bytes) -> void* {
    void* p = (void*)wsp;
    wsp += (bytes + 255) & ~(size_t)255;
    return p;
  };
  bf16* h_bf   = (bf16*)alloc((size_t)SB * DM * 2);
  bf16* g_bf   = (bf16*)alloc((size_t)SB * DM * 2);
  bf16* r_bf   = (bf16*)alloc((size_t)RB * DM * 2);
  bf16* wqT    = (bf16*)alloc((size_t)DM * DM * 2);   // transposed weights (NT B operands)
  bf16* wkT    = (bf16*)alloc((size_t)DM * DM * 2);
  bf16* wvT    = (bf16*)alloc((size_t)DM * DM * 2);
  bf16* wrT    = (bf16*)alloc((size_t)DM * DM * 2);
  bf16* wo_bf  = (bf16*)alloc((size_t)DM * DM * 2);   // wo used as-is ([h, nd] == NT layout)
  bf16* w1T    = (bf16*)alloc((size_t)DM * DI * 2);   // [DI, DM]
  bf16* w2T    = (bf16*)alloc((size_t)DI * DM * 2);   // [DM, DI]
  bf16* k_bf   = (bf16*)alloc((size_t)SB * DM * 2);
  bf16* v_bf   = (bf16*)alloc((size_t)SB * DM * 2);
  bf16* vt_bf  = (bf16*)alloc((size_t)SB * DM * 2);   // per-head transposed V [z][d][j]
  bf16* kr_bf  = (bf16*)alloc((size_t)RB * DM * 2);
  bf16* qrw_bf = (bf16*)alloc((size_t)SB * DM * 2);
  bf16* qrr_bf = (bf16*)alloc((size_t)SB * DM * 2);
  bf16* vec_bf = (bf16*)alloc((size_t)SB * DM * 2);
  float* out1_f = (float*)alloc((size_t)SB * DM * 4);
  bf16* out1_bf = (bf16*)alloc((size_t)SB * DM * 2);
  bf16* ffn1_bf = (bf16*)alloc((size_t)SB * DI * 2);
  float* ef1    = (float*)alloc((size_t)SB * H * 2 * 4);
  float* scores = (float*)alloc((size_t)32 * SS * 4);   // [b*H, S, S], 128 MB
  bf16*  prob   = (bf16*)alloc((size_t)32 * SS * 2);    // 64 MB
  float* tbuf   = (float*)alloc((size_t)SB * DI * 4);   // 32 MB reusable f32 scratch

  auto cast = [&](const float* in, bf16* o, long long n) {
    k_cast_bf16<<<(unsigned)((n + 255) / 256), 256, 0, stream>>>(in, o, n);
  };

  // ---- cast activations; transpose+cast weights ----
  cast(h, h_bf, (long long)SB * DM);
  cast(g, g_bf, (long long)SB * DM);
  cast(r, r_bf, (long long)RB * DM);
  cast(wo, wo_bf, (long long)DM * DM);
  k_transpose_cast<<<dim3(32, 32), dim3(32, 8), 0, stream>>>(wq, wqT, DM, DM);
  k_transpose_cast<<<dim3(32, 32), dim3(32, 8), 0, stream>>>(wk, wkT, DM, DM);
  k_transpose_cast<<<dim3(32, 32), dim3(32, 8), 0, stream>>>(wv, wvT, DM, DM);
  k_transpose_cast<<<dim3(32, 32), dim3(32, 8), 0, stream>>>(wr, wrT, DM, DM);
  k_transpose_cast<<<dim3(128, 32), dim3(32, 8), 0, stream>>>(ff_w1, w1T, DM, DI);
  k_transpose_cast<<<dim3(32, 128), dim3(32, 8), 0, stream>>>(ff_w2, w2T, DI, DM);

  // ---- shared K, V, K_r projections ----
  gemm_nt_wmma<0><<<dim3(8, 16, 1), 256, 0, stream>>>(
      h_bf, wkT, tbuf, SB, DM, DM, DM, DM, DM, DM, 0, 0, 0);
  cast(tbuf, k_bf, (long long)SB * DM);
  gemm_nt_wmma<0><<<dim3(8, 16, 1), 256, 0, stream>>>(
      h_bf, wvT, tbuf, SB, DM, DM, DM, DM, DM, DM, 0, 0, 0);
  cast(tbuf, v_bf, (long long)SB * DM);
  k_transpose_v<<<dim3(32, 2, 32), dim3(32, 8), 0, stream>>>(v_bf, vt_bf);
  gemm_nt_wmma<0><<<dim3(8, 32, 1), 256, 0, stream>>>(
      r_bf, wrT, tbuf, RB, DM, DM, DM, DM, DM, DM, 0, 0, 0);
  cast(tbuf, kr_bf, (long long)RB * DM);

  // ---- per stream ----
  for (int st = 0; st < 2; ++st) {
    const float* x    = st ? g : h;
    const bf16*  x_bf = st ? g_bf : h_bf;
    const float* mask = st ? mask_g : mask_h;
    float* outp = out + (long long)st * SB * DM;

    // q projection -> tbuf (f32)
    gemm_nt_wmma<0><<<dim3(8, 16, 1), 256, 0, stream>>>(
        x_bf, wqT, tbuf, SB, DM, DM, DM, DM, DM, DM, 0, 0, 0);
    k_addbias_cast<<<(unsigned)(((long long)SB * DM + 255) / 256), 256, 0, stream>>>(
        tbuf, r_w_bias, DM, qrw_bf, (long long)SB * DM);
    k_addbias_cast<<<(unsigned)(((long long)SB * DM + 255) / 256), 256, 0, stream>>>(
        tbuf, r_r_bias, DM, qrr_bf, (long long)SB * DM);
    k_ef1<<<(SB * H + 255) / 256, 256, 0, stream>>>(tbuf, r_s_bias, seg_emb, ef1, SB * H);

    // ac[z,i,j] = qrw . k^T  (batched over z = b*16+n; per-head views stride 64)
    gemm_nt_wmma<0><<<dim3(8, 8, 32), 256, 0, stream>>>(
        qrw_bf, k_bf, scores, S, S, D, S, 2048, 2048, S, 64, 64, SS);
    // bd = qrr . k_r^T with rel_shift fused into the accumulate-store
    gemm_nt_wmma<1><<<dim3(16, 8, 32), 256, 0, stream>>>(
        qrr_bf, kr_bf, scores, S, R, D, S, 2048, 2048, S, 64, 64, SS);

    // softmax (+segment term, +mask, *SCALE) -> bf16 probs
    k_softmax<<<dim3(S, 32), 256, 0, stream>>>(scores, ef1, seg_mat, mask, prob);

    // attn_vec[z] = prob[z] @ vt[z]^T -> tbuf in [i,b,n,d] layout
    gemm_nt_wmma<0><<<dim3(1, 8, 32), 256, 0, stream>>>(
        prob, vt_bf, tbuf, S, D, S, D, S, S, 2048, SS, (long long)D * S, 64);
    cast(tbuf, vec_bf, (long long)SB * DM);

    // output projection: vec @ wo^T -> tbuf
    gemm_nt_wmma<0><<<dim3(8, 16, 1), 256, 0, stream>>>(
        vec_bf, wo_bf, tbuf, SB, DM, DM, DM, DM, DM, DM, 0, 0, 0);
    // residual + LN -> out1 (f32 + bf16)
    k_resid_ln<<<SB, 256, 0, stream>>>(tbuf, x, nullptr, ln_g_at, ln_b_at,
                                       out1_f, out1_bf, DM);

    // FFN: out1 @ w1 -> gelu -> @ w2 -> +b2 + out1 -> LN -> final output
    gemm_nt_wmma<0><<<dim3(32, 16, 1), 256, 0, stream>>>(
        out1_bf, w1T, tbuf, SB, DI, DM, DI, DM, DM, DI, 0, 0, 0);
    k_bias_gelu_cast<<<(unsigned)(((long long)SB * DI + 255) / 256), 256, 0, stream>>>(
        tbuf, ff_b1, DI, ffn1_bf, (long long)SB * DI);
    gemm_nt_wmma<0><<<dim3(8, 16, 1), 256, 0, stream>>>(
        ffn1_bf, w2T, tbuf, SB, DM, DI, DM, DI, DI, DM, 0, 0, 0);
    k_resid_ln<<<SB, 256, 0, stream>>>(tbuf, out1_f, ff_b2, ln_g_ff, ln_b_ff,
                                       outp, nullptr, DM);
  }
}
